// MultiHeadAttention_77455440216268
// MI455X (gfx1250) — matrix operand order repaired and checked by reading, not compiled, not run on hardware
//
#include <hip/hip_runtime.h>

#define DEVINL __device__ __forceinline__

typedef __attribute__((ext_vector_type(16))) __bf16 v16bf;
typedef __attribute__((ext_vector_type(8)))  __bf16 v8bf;
typedef __attribute__((ext_vector_type(8)))  float  v8f;

constexpr int Bq = 2, Tq = 2048, Cq = 1024, Hq = 16, Dq = 64;

DEVINL unsigned short f2bfbits(float f) {
    unsigned u = __builtin_bit_cast(unsigned, f);
    u += 0x7FFFu + ((u >> 16) & 1u);          // round-to-nearest-even
    return (unsigned short)(u >> 16);
}

DEVINL v8f vzero() {
    v8f z;
#pragma unroll
    for (int i = 0; i < 8; ++i) z[i] = 0.f;
    return z;
}

// 16 contiguous bf16 (two aligned 16B halves)
DEVINL v16bf ld16(const unsigned short* p) {
    v8bf lo = *reinterpret_cast<const v8bf*>(p);
    v8bf hi = *reinterpret_cast<const v8bf*>(p + 8);
    return __builtin_shufflevector(lo, hi, 0,1,2,3,4,5,6,7,8,9,10,11,12,13,14,15);
}
// two separate 8-element bf16 chunks (A-operand: K 0-7 / 16-23 halves)
DEVINL v16bf ld8x2(const unsigned short* p0, const unsigned short* p1) {
    v8bf lo = *reinterpret_cast<const v8bf*>(p0);
    v8bf hi = *reinterpret_cast<const v8bf*>(p1);
    return __builtin_shufflevector(lo, hi, 0,1,2,3,4,5,6,7,8,9,10,11,12,13,14,15);
}

DEVINL v8f wmma_bf16(v16bf a, v16bf b, v8f c) {
    return __builtin_amdgcn_wmma_f32_16x16x32_bf16(false, a, false, b, (short)0, c,
                                                   false, false);
}

// async 32B global -> LDS copy (bypasses VGPRs, tracked by ASYNCcnt).
// NOTE: inst offset applies to BOTH lds and global addresses on this op.
DEVINL void async_cp32(unsigned lds_off, unsigned long long gaddr) {
    asm volatile(
        "global_load_async_to_lds_b128 %0, %1, off\n\t"
        "global_load_async_to_lds_b128 %0, %1, off offset:16"
        :: "v"(lds_off), "v"(gaddr) : "memory");
}

// ---------------- prep: fp32 -> bf16 ----------------
__global__ void k_cvt(const float* __restrict__ src, unsigned short* __restrict__ dst, int n) {
    int i = blockIdx.x * blockDim.x + threadIdx.x;
    if (i < n) dst[i] = f2bfbits(src[i]);
}

// dst[b][c][r] = bf16(src[b][r][c]); batch over blockIdx.y
__global__ void k_transpose_cvt(const float* __restrict__ src, unsigned short* __restrict__ dst,
                                int rows, int cols) {
    int bb = blockIdx.y;
    size_t base = (size_t)bb * rows * cols;
    int i = blockIdx.x * blockDim.x + threadIdx.x;
    if (i < rows * cols) {
        int r = i / cols, c = i % cols;
        dst[base + (size_t)c * rows + r] = f2bfbits(src[base + i]);
    }
}

// ---------------- pass 1: fused QKV projection ----------------
__global__ __launch_bounds__(32) void k_qkv(const unsigned short* __restrict__ xb,
                                            const unsigned short* __restrict__ wqt,
                                            const unsigned short* __restrict__ wkt,
                                            const unsigned short* __restrict__ wvt,
                                            unsigned short* __restrict__ Qo,
                                            unsigned short* __restrict__ Ko,
                                            unsigned short* __restrict__ Vto) {
    int lane = threadIdx.x;
    int w = blockIdx.x;
    int mat = w % 3;  w /= 3;        // 0=Q 1=K 2=V
    int h   = w % Hq; w /= Hq;
    int b   = w / (Tq / 16);
    int t0  = (w % (Tq / 16)) * 16;

    const unsigned short* wt = (mat == 0 ? wqt : (mat == 1 ? wkt : wvt)) + (size_t)h * Dq * Cq;

    int row = lane & 15, hi = lane >> 4, col = row;
    const unsigned short* arow = xb + ((size_t)b * Tq + t0 + row) * Cq + hi * 16;
    const unsigned short* brow = wt + (size_t)col * Cq + hi * 16;

    v8f acc[4];
#pragma unroll
    for (int j = 0; j < 4; ++j) acc[j] = vzero();

#pragma unroll 2
    for (int k0 = 0; k0 < Cq; k0 += 32) {
        __builtin_prefetch(arow + k0 + 128, 0, 1);
        v16bf a = ld16(arow + k0);
#pragma unroll
        for (int j = 0; j < 4; ++j) {
            v16bf bb = ld16(brow + (size_t)j * 16 * Cq + k0);
            acc[j] = wmma_bf16(a, bb, acc[j]);
        }
    }

    size_t hb = (size_t)b * Hq + h;
#pragma unroll
    for (int j = 0; j < 4; ++j) {
        int d = j * 16 + col;
#pragma unroll
        for (int r = 0; r < 8; ++r) {
            int m = r + hi * 8;
            unsigned short val = f2bfbits(acc[j][r]);
            if (mat == 2) {
                Vto[(hb * Dq + d) * Tq + (t0 + m)] = val;          // V transposed [B,H,D,T]
            } else {
                unsigned short* dst = (mat == 0 ? Qo : Ko);
                dst[(hb * Tq + (t0 + m)) * Dq + d] = val;          // [B,H,T,D]
            }
        }
    }
}

// ---------------- pass 2: flash attention ----------------
// 4 waves = 64 consecutive queries of one (b,h). K/V blocks are staged into
// double-buffered LDS with async direct-to-LDS copies; block i+1's fetch is
// issued before waiting on block i (ASYNCcnt in-order retire), hiding latency.
__global__ __launch_bounds__(128) void k_attn(const unsigned short* __restrict__ Q,
                                              const unsigned short* __restrict__ K,
                                              const unsigned short* __restrict__ Vt,
                                              unsigned short* __restrict__ attn) {
    __shared__ __align__(16) unsigned short Kl[2][32 * 64];   // [buf][key][d]
    __shared__ __align__(16) unsigned short Vl[2][64 * 32];   // [buf][d][key]
    __shared__ __align__(16) unsigned short Pl[4 * 16 * 32];

    int tid  = threadIdx.x;
    int lane = tid & 31;
    int wv   = tid >> 5;

    int w = blockIdx.x;
    int base = (w % (Tq / 64)) * 64;  w /= (Tq / 64);
    int h = w % Hq;
    int b = w / Hq;
    int t0 = base + wv * 16;

    size_t hb = (size_t)b * Hq + h;
    const unsigned short* Qp = Q  + hb * Tq * Dq;
    const unsigned short* Kp = K  + hb * Tq * Dq;
    const unsigned short* Vp = Vt + hb * Dq * Tq;

    int row = lane & 15, hi = lane >> 4, col = row;
    unsigned short* Pw = &Pl[wv * 16 * 32];

    // per-thread staging slots (32B each of K and V^T per thread per block)
    int kr = tid >> 2, kc = (tid & 3) * 16;                    // K: 32 rows x 64
    int vr = tid >> 1, vc = (tid & 1) * 16;                    // V^T: 64 rows x 32
    unsigned klo[2], vlo[2];
#pragma unroll
    for (int p = 0; p < 2; ++p) {
        klo[p] = (unsigned)(uintptr_t)&Kl[p][kr * 64 + kc];
        vlo[p] = (unsigned)(uintptr_t)&Vl[p][vr * 32 + vc];
    }
    auto issue = [&](int s0, int buf) {
        async_cp32(klo[buf], (unsigned long long)(Kp + (size_t)(s0 + kr) * Dq + kc));
        async_cp32(vlo[buf], (unsigned long long)(Vp + (size_t)vr * Tq + s0 + vc));
    };

    v16bf aQ[2];
#pragma unroll
    for (int kk = 0; kk < 2; ++kk) {
        const unsigned short* p = Qp + (size_t)(t0 + row) * Dq + kk * 32 + hi * 16;
        aQ[kk] = ld16(p);
    }

    v8f o[4];
#pragma unroll
    for (int j = 0; j < 4; ++j) o[j] = vzero();
    float M[8], L[8];
#pragma unroll
    for (int r = 0; r < 8; ++r) { M[r] = -__builtin_inff(); L[r] = 0.f; }
    const float scale = 0.125f; // 1/sqrt(64)

    int nb = (base + 64) >> 5;          // group-uniform block count
    issue(0, 0);
    for (int i = 0; i < nb; ++i) {
        int cur = i & 1;
        int s0  = i * 32;
        // prefetch next block into the other buffer (dummy re-read of block 0 on last)
        int nxt = (i + 1 < nb) ? s0 + 32 : 0;
        issue(nxt, cur ^ 1);
        // wait for current block only (4 async ops of next batch may stay pending)
        asm volatile("s_wait_asynccnt 0x4" ::: "memory");
        __syncthreads();

        const unsigned short* Kb = Kl[cur];
        const unsigned short* Vb = Vl[cur];

        // ---- S = scale * Q K^T ----
        v8f sv[2]; sv[0] = vzero(); sv[1] = vzero();
#pragma unroll
        for (int kk = 0; kk < 2; ++kk) {
#pragma unroll
            for (int j = 0; j < 2; ++j) {
                v16bf bb = ld16(&Kb[(j * 16 + col) * 64 + kk * 32 + hi * 16]);
                sv[j] = wmma_bf16(aQ[kk], bb, sv[j]);
            }
        }

        // ---- online softmax (rows live across 16-lane halves) ----
#pragma unroll
        for (int r = 0; r < 8; ++r) {
            int m = r + hi * 8;
            int t = t0 + m;
            float v0 = sv[0][r] * scale;
            float v1 = sv[1][r] * scale;
            if (s0 + col > t)      v0 = -__builtin_inff();
            if (s0 + 16 + col > t) v1 = -__builtin_inff();
            float rm = fmaxf(v0, v1);
#pragma unroll
            for (int x = 1; x < 16; x <<= 1) rm = fmaxf(rm, __shfl_xor(rm, x, 32));
            float Mn   = fmaxf(M[r], rm);
            float corr = __expf(M[r] - Mn);
            float p0   = __expf(v0 - Mn);
            float p1   = __expf(v1 - Mn);
            float rs = p0 + p1;
#pragma unroll
            for (int x = 1; x < 16; x <<= 1) rs += __shfl_xor(rs, x, 32);
            L[r] = L[r] * corr + rs;
            M[r] = Mn;
#pragma unroll
            for (int j = 0; j < 4; ++j) o[j][r] *= corr;
            Pw[m * 32 + col]      = f2bfbits(p0);
            Pw[m * 32 + col + 16] = f2bfbits(p1);
        }
        __syncthreads();   // P visible; waves stay in lockstep

        // ---- O += P * V ----
        const unsigned short* pp = &Pw[row * 32 + hi * 16];
        v16bf aP = ld16(pp);
#pragma unroll
        for (int j = 0; j < 4; ++j) {
            v16bf bb = ld16(&Vb[(j * 16 + col) * 32 + hi * 16]);
            o[j] = wmma_bf16(aP, bb, o[j]);
        }
        __syncthreads();   // reads of buf[cur] done before it is re-staged
    }

#pragma unroll
    for (int r = 0; r < 8; ++r) {
        float inv = 1.f / L[r];
        int m = r + hi * 8;
        size_t rowb = ((size_t)b * Tq + t0 + m) * Cq + (size_t)h * Dq;
#pragma unroll
        for (int j = 0; j < 4; ++j)
            attn[rowb + j * 16 + col] = f2bfbits(o[j][r] * inv);
    }
}

// ---------------- pass 3: output projection + bias ----------------
__global__ __launch_bounds__(32) void k_proj(const unsigned short* __restrict__ attn,
                                             const unsigned short* __restrict__ wpt,
                                             const float* __restrict__ bias,
                                             float* __restrict__ out) {
    int lane = threadIdx.x;
    int w = blockIdx.x;
    int n0 = (w % (Cq / 64)) * 64;
    int tm = w / (Cq / 64);
    int b  = tm / (Tq / 16);
    int t0 = (tm % (Tq / 16)) * 16;

    int row = lane & 15, hi = lane >> 4, col = row;
    const unsigned short* Ab = attn + ((size_t)b * Tq + t0 + row) * Cq + hi * 16;
    const unsigned short* Bb = wpt + (size_t)(n0 + col) * Cq + hi * 16;

    v8f acc[4];
#pragma unroll
    for (int j = 0; j < 4; ++j) acc[j] = vzero();

#pragma unroll 2
    for (int k0 = 0; k0 < Cq; k0 += 32) {
        __builtin_prefetch(Ab + k0 + 128, 0, 1);
        v16bf a = ld16(Ab + k0);
#pragma unroll
        for (int j = 0; j < 4; ++j) {
            v16bf bb = ld16(Bb + (size_t)j * 16 * Cq + k0);
            acc[j] = wmma_bf16(a, bb, acc[j]);
        }
    }
#pragma unroll
    for (int j = 0; j < 4; ++j) {
        float bv = bias[n0 + j * 16 + col];
#pragma unroll
        for (int r = 0; r < 8; ++r) {
            int m = r + hi * 8;
            out[((size_t)b * Tq + t0 + m) * Cq + n0 + j * 16 + col] = acc[j][r] + bv;
        }
    }
}

extern "C" void kernel_launch(void* const* d_in, const int* in_sizes, int n_in,
                              void* d_out, int out_size, void* d_ws, size_t ws_size,
                              hipStream_t stream) {
    const float* x  = (const float*)d_in[0];
    const float* Wq = (const float*)d_in[1];
    const float* Wk = (const float*)d_in[2];
    const float* Wv = (const float*)d_in[3];
    const float* Wp = (const float*)d_in[4];
    const float* bp = (const float*)d_in[5];
    float* out = (float*)d_out;

    char* ws = (char*)d_ws;
    auto alloc = [&](size_t bytes) -> void* {
        void* p = ws;
        ws += (bytes + 255) & ~(size_t)255;
        return p;
    };
    unsigned short* xb   = (unsigned short*)alloc((size_t)Bq * Tq * Cq * 2);
    unsigned short* wqt  = (unsigned short*)alloc((size_t)Hq * Dq * Cq * 2);
    unsigned short* wkt  = (unsigned short*)alloc((size_t)Hq * Dq * Cq * 2);
    unsigned short* wvt  = (unsigned short*)alloc((size_t)Hq * Dq * Cq * 2);
    unsigned short* wpt  = (unsigned short*)alloc((size_t)Cq * Cq * 2);
    unsigned short* Qb   = (unsigned short*)alloc((size_t)Bq * Hq * Tq * Dq * 2);
    unsigned short* Kb   = (unsigned short*)alloc((size_t)Bq * Hq * Tq * Dq * 2);
    unsigned short* Vtb  = (unsigned short*)alloc((size_t)Bq * Hq * Dq * Tq * 2);
    unsigned short* attn = (unsigned short*)alloc((size_t)Bq * Tq * Cq * 2);

    int n = Bq * Tq * Cq;
    k_cvt<<<(n + 255) / 256, 256, 0, stream>>>(x, xb, n);

    dim3 gw((Cq * Dq + 255) / 256, Hq);
    k_transpose_cvt<<<gw, 256, 0, stream>>>(Wq, wqt, Cq, Dq);
    k_transpose_cvt<<<gw, 256, 0, stream>>>(Wk, wkt, Cq, Dq);
    k_transpose_cvt<<<gw, 256, 0, stream>>>(Wv, wvt, Cq, Dq);
    dim3 gp((Cq * Cq + 255) / 256, 1);
    k_transpose_cvt<<<gp, 256, 0, stream>>>(Wp, wpt, Cq, Cq);

    k_qkv<<<(Bq * Tq / 16) * Hq * 3, 32, 0, stream>>>(xb, wqt, wkt, wvt, Qb, Kb, Vtb);
    k_attn<<<Bq * Hq * (Tq / 64), 128, 0, stream>>>(Qb, Kb, Vtb, attn);
    k_proj<<<(Bq * Tq / 16) * (Cq / 64), 32, 0, stream>>>(attn, wpt, bp, out);
}
